// CrossAttention_15195594293827
// MI455X (gfx1250) — matrix operand order repaired and checked by reading, not compiled, not run on hardware
//
#include <hip/hip_runtime.h>

// ---------------------------------------------------------------------------
// Problem constants: B=8, LQ=LK=1024, H1=H2=768, 3 heads.
// ---------------------------------------------------------------------------
typedef __bf16 v16bf __attribute__((ext_vector_type(16)));
typedef float  v8f   __attribute__((ext_vector_type(8)));

__device__ __forceinline__ v8f wmma_bf16(v16bf a, v16bf b, v8f c) {
  // (neg_a, A, neg_b, B, c_mod, C, reuse_a, reuse_b)
  return __builtin_amdgcn_wmma_f32_16x16x32_bf16(false, a, false, b, (short)0, c,
                                                 false, false);
}

// Async direct copy global -> LDS (16 bytes per lane), tracked by ASYNCcnt.
// ldsOff is the wave-relative LDS byte offset (= low 32 bits of a generic LDS
// pointer, per CDNA5 ISA 10.2 aperture mapping).
__device__ __forceinline__ void async_copy16(unsigned ldsOff,
                                             const void* gaddr) {
  const unsigned long long ga = (unsigned long long)(size_t)gaddr;
  asm volatile("global_load_async_to_lds_b128 %0, %1, off"
               :
               : "v"(ldsOff), "v"(ga)
               : "memory");
}

__device__ __forceinline__ void wait_async() {
  asm volatile("s_wait_asynccnt 0x0" ::: "memory");
}

// A fragment: 16x32 bf16 tile; element (m,k) at src[m*ld + k] (row-major).
// VGPR j<4 : K = 16*half + 2j ; VGPR 4+j : K = 16*half + 8 + 2j ; M = lane&15.
__device__ __forceinline__ v16bf load_frag_a(const __bf16* src, int ld, int lane) {
  const int half = lane >> 4, r = lane & 15;
  const __bf16* row = src + r * ld;
  v16bf a{};
#pragma unroll
  for (int j = 0; j < 4; ++j) {
    const int k0 = half * 16 + 2 * j;
    a[2 * j]     = row[k0];
    a[2 * j + 1] = row[k0 + 1];
    const int k1 = half * 16 + 8 + 2 * j;
    a[8 + 2 * j]     = row[k1];
    a[8 + 2 * j + 1] = row[k1 + 1];
  }
  return a;
}

// B fragment: 32(k) x 16(n); element (k,n) at src[n*ld + k] ("n-major" storage
// so consecutive-k pairs are contiguous). VGPR j: K = 16*half + 2j ; N = lane&15.
__device__ __forceinline__ v16bf load_frag_b(const __bf16* src, int ld, int lane) {
  const int half = lane >> 4, c = lane & 15;
  const __bf16* col = src + c * ld;
  v16bf b{};
#pragma unroll
  for (int j = 0; j < 8; ++j) {
    const int k = 16 * half + 2 * j;
    b[2 * j]     = col[k];
    b[2 * j + 1] = col[k + 1];
  }
  return b;
}

// ---------------------------------------------------------------------------
// Generic tiled GEMM: C[M,N] = A[M,K] * W[K,N] + bias[N]
// Block tile 128x128, BK=32, 256 threads = 8 waves arranged 4(M) x 2(N);
// each wave: 2x4 WMMA tiles (32x64). W is fp32, converted to bf16 into LDS
// transposed (n-major). A is fp32 or bf16 per template flag.
// ---------------------------------------------------------------------------
template <bool A_BF16, bool OUT_BF16>
__device__ __forceinline__ void gemm_block(const void* Ap, const float* W,
                                           const float* bias, void* Cp,
                                           int N, int K) {
  __shared__ __bf16 As[128 * 36];   // row-major, stride 36
  __shared__ __bf16 Bst[128 * 36];  // n-major (transposed), stride 36

  const int tid  = threadIdx.x;
  const int lane = tid & 31;
  const int w    = tid >> 5;
  const int wm   = w & 3;   // 0..3
  const int wn   = w >> 2;  // 0..1
  const int m0   = blockIdx.y * 128;
  const int n0   = blockIdx.x * 128;

  v8f acc[2][4] = {};

  for (int kk = 0; kk < K; kk += 32) {
    __syncthreads();
    // ---- load A tile (128 x 32) ----
    if (A_BF16) {
      const __bf16* Ab = (const __bf16*)Ap;
      const int g = tid & 3, rb = tid >> 2;  // 8-elem col group, rows 0..63
#pragma unroll
      for (int rr = 0; rr < 2; ++rr) {
        const int r = rb + rr * 64;
        // straight bf16 copy: use the async global->LDS direct path
        async_copy16((unsigned)(size_t)(As + r * 36 + g * 8),
                     Ab + (size_t)(m0 + r) * K + kk + g * 8);
      }
    } else {
      const float* Af = (const float*)Ap;
      const int g = tid & 7, rb = tid >> 3;  // 4-elem col group, rows 0..31
#pragma unroll
      for (int rr = 0; rr < 4; ++rr) {
        const int r = rb + rr * 32;
        float4 d = *(const float4*)(Af + (size_t)(m0 + r) * K + kk + g * 4);
        As[r * 36 + g * 4 + 0] = (__bf16)d.x;
        As[r * 36 + g * 4 + 1] = (__bf16)d.y;
        As[r * 36 + g * 4 + 2] = (__bf16)d.z;
        As[r * 36 + g * 4 + 3] = (__bf16)d.w;
      }
    }
    // ---- load W tile (32 x 128) transposed into Bst[n][k] ----
    {
      const int gn = tid & 31, kb = tid >> 5;  // 4-col group, k rows 0..7
#pragma unroll
      for (int rr = 0; rr < 4; ++rr) {
        const int kr = kb + rr * 8;
        float4 d = *(const float4*)(W + (size_t)(kk + kr) * N + n0 + gn * 4);
        Bst[(gn * 4 + 0) * 36 + kr] = (__bf16)d.x;
        Bst[(gn * 4 + 1) * 36 + kr] = (__bf16)d.y;
        Bst[(gn * 4 + 2) * 36 + kr] = (__bf16)d.z;
        Bst[(gn * 4 + 3) * 36 + kr] = (__bf16)d.w;
      }
    }
    if (A_BF16) wait_async();
    __syncthreads();
    // ---- compute ----
    v16bf afr[2], bfr[4];
#pragma unroll
    for (int mt = 0; mt < 2; ++mt)
      afr[mt] = load_frag_a(&As[(wm * 32 + mt * 16) * 36], 36, lane);
#pragma unroll
    for (int nt = 0; nt < 4; ++nt)
      bfr[nt] = load_frag_b(&Bst[(wn * 64 + nt * 16) * 36], 36, lane);
#pragma unroll
    for (int mt = 0; mt < 2; ++mt)
#pragma unroll
      for (int nt = 0; nt < 4; ++nt)
        acc[mt][nt] = wmma_bf16(afr[mt], bfr[nt], acc[mt][nt]);
  }

  // ---- epilogue: bias add + store ----
  const int half = lane >> 4, c = lane & 15;
#pragma unroll
  for (int mt = 0; mt < 2; ++mt)
#pragma unroll
    for (int nt = 0; nt < 4; ++nt) {
      const int col = n0 + wn * 64 + nt * 16 + c;
      const float bv = bias[col];
#pragma unroll
      for (int r = 0; r < 8; ++r) {
        const int row = m0 + wm * 32 + mt * 16 + half * 8 + r;
        const float v = acc[mt][nt][r] + bv;
        if (OUT_BF16)
          ((__bf16*)Cp)[(size_t)row * N + col] = (__bf16)v;
        else
          ((float*)Cp)[(size_t)row * N + col] = v;
      }
    }
}

// ---------------------------------------------------------------------------
// Kernel 1: Q/K/V projections. grid.z = 0..8 selects {Q,K,V} x head.
// X[8192,768] (fp32) * W[768,768] + b -> out bf16 [head][8192][768]
// ---------------------------------------------------------------------------
__global__ __launch_bounds__(256) void qkv_kernel(
    const float* t1, const float* t2, const float* Wq, const float* bq,
    const float* Wk, const float* bk, const float* Wv, const float* bv,
    void* Qb, void* Kb, void* Vb) {
  const int p = blockIdx.z;
  const int typ = p / 3, h = p % 3;
  const size_t woff = (size_t)h * 768 * 768;
  const size_t ooff = (size_t)h * 8192 * 768;
  const float* X;
  const float* W;
  const float* bia;
  __bf16* O;
  if (typ == 0) { X = t1; W = Wq + woff; bia = bq + h * 768; O = (__bf16*)Qb + ooff; }
  else if (typ == 1) { X = t2; W = Wk + woff; bia = bk + h * 768; O = (__bf16*)Kb + ooff; }
  else { X = t2; W = Wv + woff; bia = bv + h * 768; O = (__bf16*)Vb + ooff; }
  gemm_block<false, true>(X, W, bia, O, 768, 768);
}

// ---------------------------------------------------------------------------
// Kernel 2: flash attention. Block = (32 queries) x (head,batch).
// 256 threads = 8 waves. Per 64-key iteration:
//   S(32x64) = Q * K^T   (8 waves, 1 tile each, 24 WMMA over K=768)
//   online softmax (rows), P bf16
//   ctx(32x768) += P * V (each wave: 2x6 tiles x 2 k-steps)
// Q and K tiles staged via GLOBAL_LOAD_ASYNC_TO_LDS_B128 (ASYNCcnt);
// V is transposed during staging (manual path).
// ReLU(ctx / l) stored bf16 into multi[B*LQ][2304] at feature offset h*768.
// ---------------------------------------------------------------------------
constexpr int QLD = 776;  // 768 + 8 pad (bf16 elems)
constexpr int VLD = 68;   // 64 + 4 pad
constexpr int SLD = 68;
constexpr size_t ATTN_SMEM =
    (size_t)(32 * QLD + 64 * QLD + 768 * VLD + 32 * SLD) * 2 +
    (size_t)(32 * SLD + 96) * 4;  // = 266,880 bytes (< 320 KB WGP LDS)

__global__ __launch_bounds__(256) void attn_kernel(const void* Qb_, const void* Kb_,
                                                   const void* Vb_, void* Mb_) {
  extern __shared__ char smem[];
  __bf16* Qs = (__bf16*)smem;          // [32][QLD]
  __bf16* Ks = Qs + 32 * QLD;          // [64][QLD]
  __bf16* Vt = Ks + 64 * QLD;          // [768][VLD]  (transposed: [feat][key])
  __bf16* Ps = Vt + 768 * VLD;         // [32][SLD]
  float*  Sf = (float*)(Ps + 32 * SLD);  // [32][SLD]
  float*  mst = Sf + 32 * SLD;           // [32]
  float*  lst = mst + 32;                // [32]
  float*  ssc = lst + 32;                // [32]

  const int tid  = threadIdx.x;
  const int lane = tid & 31;
  const int w    = tid >> 5;
  const int q0   = blockIdx.x * 32;
  const int b    = blockIdx.y;
  const int h    = blockIdx.z;

  const size_t hb = (size_t)(h * 8 + b) * 1024;
  const __bf16* Qg = (const __bf16*)Qb_ + (hb + q0) * 768;
  const __bf16* Kg = (const __bf16*)Kb_ + hb * 768;
  const __bf16* Vg = (const __bf16*)Vb_ + hb * 768;
  __bf16* Mb = (__bf16*)Mb_;

  // stage Q tile (32x768 bf16) into LDS via async direct copy
  for (int i = 0; i < 12; ++i) {
    const int e8 = tid + i * 256;       // 3072 groups of 8 elems
    const int row = e8 / 96, cg = e8 % 96;
    async_copy16((unsigned)(size_t)(Qs + row * QLD + cg * 8),
                 Qg + (size_t)row * 768 + cg * 8);
  }
  if (tid < 32) { mst[tid] = -INFINITY; lst[tid] = 0.f; }

  v8f acc[2][6] = {};
  const int stm = w >> 2, stn = w & 3;  // this wave's S tile (2x4 grid)
  const int colb = w * 96;              // this wave's ctx column base

  for (int kb = 0; kb < 1024; kb += 64) {
    __syncthreads();
    // ---- stage K (async, row-major) and V (transposed, manual) blocks ----
    for (int i = 0; i < 24; ++i) {
      const int e8 = tid + i * 256;     // 6144 groups: row=key 0..63, cg 0..95
      const int row = e8 / 96, cg = e8 % 96;
      async_copy16((unsigned)(size_t)(Ks + row * QLD + cg * 8),
                   Kg + (size_t)(kb + row) * 768 + cg * 8);
      float4 dv = *(const float4*)(Vg + (size_t)(kb + row) * 768 + cg * 8);
      const __bf16* pv = (const __bf16*)&dv;
#pragma unroll
      for (int e = 0; e < 8; ++e) Vt[(cg * 8 + e) * VLD + row] = pv[e];
    }
    wait_async();                        // Q (first iter) + K copies complete
    __syncthreads();

    // ---- S = Q K^T : wave computes 16x16 tile (stm, stn) over K=768 ----
    v8f s = {};
#pragma unroll 4
    for (int kt = 0; kt < 24; ++kt) {
      v16bf a  = load_frag_a(&Qs[(stm * 16) * QLD + kt * 32], QLD, lane);
      v16bf bb = load_frag_b(&Ks[(stn * 16) * QLD + kt * 32], QLD, lane);
      s = wmma_bf16(a, bb, s);
    }
    {
      const int half = lane >> 4, c = lane & 15;
#pragma unroll
      for (int r = 0; r < 8; ++r)
        Sf[(stm * 16 + half * 8 + r) * SLD + stn * 16 + c] = s[r];
    }
    __syncthreads();

    // ---- online softmax (one thread per query row) ----
    if (tid < 32) {
      const float mold = mst[tid], lold = lst[tid];
      float mb = -INFINITY;
#pragma unroll 8
      for (int c = 0; c < 64; ++c) mb = fmaxf(mb, Sf[tid * SLD + c]);
      const float mnew = fmaxf(mold, mb);
      const float sc = __expf(mold - mnew);
      float sum = 0.f;
#pragma unroll 8
      for (int c = 0; c < 64; ++c) {
        const float pv = __expf(Sf[tid * SLD + c] - mnew);
        sum += pv;
        Ps[tid * SLD + c] = (__bf16)pv;
      }
      mst[tid] = mnew;
      lst[tid] = lold * sc + sum;
      ssc[tid] = sc;
    }
    __syncthreads();

    // ---- rescale accumulators, then ctx += P * V ----
    {
      const int half = lane >> 4;
#pragma unroll
      for (int mt = 0; mt < 2; ++mt)
#pragma unroll
        for (int r = 0; r < 8; ++r) {
          const float sc = ssc[mt * 16 + half * 8 + r];
#pragma unroll
          for (int nt = 0; nt < 6; ++nt) acc[mt][nt][r] *= sc;
        }
    }
#pragma unroll
    for (int ks = 0; ks < 2; ++ks) {
      v16bf ap0 = load_frag_a(Ps + 0 * SLD + ks * 32, SLD, lane);
      v16bf ap1 = load_frag_a(Ps + 16 * SLD + ks * 32, SLD, lane);
#pragma unroll
      for (int nt = 0; nt < 6; ++nt) {
        v16bf bv = load_frag_b(Vt + (size_t)(colb + nt * 16) * VLD + ks * 32,
                               VLD, lane);
        acc[0][nt] = wmma_bf16(ap0, bv, acc[0][nt]);
        acc[1][nt] = wmma_bf16(ap1, bv, acc[1][nt]);
      }
    }
  }
  __syncthreads();

  // ---- epilogue: ctx/l, ReLU, store bf16 into multi[B*LQ][2304] ----
  {
    const int half = lane >> 4, c = lane & 15;
#pragma unroll
    for (int mt = 0; mt < 2; ++mt)
#pragma unroll
      for (int r = 0; r < 8; ++r) {
        const int m = mt * 16 + half * 8 + r;
        const float linv = 1.f / lst[m];
        const size_t rowoff = (size_t)(b * 1024 + q0 + m) * 2304 + h * 768;
#pragma unroll
        for (int nt = 0; nt < 6; ++nt) {
          float v = acc[mt][nt][r] * linv;
          v = v > 0.f ? v : 0.f;
          Mb[rowoff + colb + nt * 16 + c] = (__bf16)v;
        }
      }
  }
}

// ---------------------------------------------------------------------------
// Kernel 3: out[8192,768] = multi_bf16[8192,2304] * Wp[2304,768] + bp
// ---------------------------------------------------------------------------
__global__ __launch_bounds__(256) void proj_kernel(const void* Mb, const float* Wp,
                                                   const float* bp, float* out) {
  gemm_block<true, false>(Mb, Wp, bp, out, 768, 2304);
}

// ---------------------------------------------------------------------------
extern "C" void kernel_launch(void* const* d_in, const int* in_sizes, int n_in,
                              void* d_out, int out_size, void* d_ws, size_t ws_size,
                              hipStream_t stream) {
  (void)in_sizes; (void)n_in; (void)out_size; (void)ws_size;
  const float* t1 = (const float*)d_in[0];
  const float* t2 = (const float*)d_in[1];
  const float* Wq = (const float*)d_in[2];
  const float* bq = (const float*)d_in[3];
  const float* Wk = (const float*)d_in[4];
  const float* bk = (const float*)d_in[5];
  const float* Wv = (const float*)d_in[6];
  const float* bv = (const float*)d_in[7];
  const float* Wp = (const float*)d_in[8];
  const float* bp = (const float*)d_in[9];
  float* out = (float*)d_out;

  // workspace layout (bf16): Qb, Kb, Vb each [3][8192][768]; multi [8192][2304]
  char* ws = (char*)d_ws;
  const size_t QKV_BYTES = (size_t)3 * 8192 * 768 * 2;  // 37,748,736
  void* Qb = ws;
  void* Kb = ws + QKV_BYTES;
  void* Vb = ws + 2 * QKV_BYTES;
  void* Mb = ws + 3 * QKV_BYTES;

  const dim3 blk(256);
  // 1) nine projection GEMMs (3 heads x {Q,K,V})
  qkv_kernel<<<dim3(6, 64, 9), blk, 0, stream>>>(t1, t2, Wq, bq, Wk, bk, Wv, bv,
                                                 Qb, Kb, Vb);
  // 2) flash attention: 32-query blocks x batch x head
  attn_kernel<<<dim3(32, 8, 3), blk, ATTN_SMEM, stream>>>(Qb, Kb, Vb, Mb);
  // 3) output projection
  proj_kernel<<<dim3(6, 64), blk, 0, stream>>>(Mb, Wp, bp, out);
}
